// LocalLearningUnit_27049704030600
// MI455X (gfx1250) — matrix operand order repaired and checked by reading, not compiled, not run on hardware
//
#include <hip/hip_runtime.h>
#include <hip/hip_bf16.h>
#include <stdint.h>

typedef __attribute__((ext_vector_type(16))) __bf16 v16bf;
typedef __attribute__((ext_vector_type(8)))  float  v8f;

union Frag16 { v16bf v; uint32_t u[8]; };

__device__ __forceinline__ unsigned short f2bf(float f) {
    unsigned int u = __float_as_uint(f);
    u += 0x7FFFu + ((u >> 16) & 1u);      // round-to-nearest-even
    return (unsigned short)(u >> 16);
}

// ---------------- conversion / setup kernels ----------------

__global__ __launch_bounds__(256) void convx_kernel(const float* __restrict__ x,
                                                    unsigned short* __restrict__ xb, int n) {
    int i = blockIdx.x * 256 + threadIdx.x;
    if (i < n) xb[i] = f2bf(x[i]);
}

// src: [batches][128 k][128 n] f32 row-major  ->  dst: [batches][128 n][128 k] bf16
__global__ __launch_bounds__(256) void convw_kernel(const float* __restrict__ src,
                                                    unsigned short* __restrict__ dst, int batches) {
    int i = blockIdx.x * 256 + threadIdx.x;
    int total = batches << 14;
    if (i >= total) return;
    int b = i >> 14, rem = i & 16383, k = rem >> 7, n = rem & 127;
    dst[((size_t)b << 14) + n * 128 + k] = f2bf(src[i]);
}

__global__ __launch_bounds__(256) void zero_kernel(float* __restrict__ p, int n) {
    int i = blockIdx.x * 256 + threadIdx.x;
    if (i < n) p[i] = 0.0f;
}

__global__ __launch_bounds__(256) void count_kernel(const int* __restrict__ et,
                                                    const int* __restrict__ dst,
                                                    float* __restrict__ cnt, int E, int N) {
    int e = blockIdx.x * 256 + threadIdx.x;
    if (e < E) unsafeAtomicAdd(&cnt[(size_t)et[e] * N + dst[e]], 1.0f);
}

// ---------------- bf16 WMMA GEMM: C[N][128] = A[N][128] @ B[128][128] (+bias) ----------------
// A: bf16 row-major [row][k]. Bt: bf16 transposed [n][k]. One wave -> 16 rows x 128 cols.

__global__ __launch_bounds__(256) void rgcn_gemm_bf16(const unsigned short* __restrict__ A,
                                                      const unsigned short* __restrict__ Bt,
                                                      const float* __restrict__ bias,
                                                      float* __restrict__ C, int nrows) {
    const int lane = threadIdx.x & 31;
    const int wave = threadIdx.x >> 5;
    const int row_base = (blockIdx.x * 8 + wave) * 16;
    if (row_base >= nrows) return;

    const int m  = lane & 15;      // A row-in-tile == C column-in-tile
    const int hi = lane >> 4;      // lane half

    int row  = row_base + m;
    int lrow = row < nrows ? row : (nrows - 1);
    const uint32_t* arow = (const uint32_t*)(A + (size_t)lrow * 128);

    v8f acc[8] = {};

    #pragma unroll
    for (int ks = 0; ks < 4; ks++) {
        const int k0 = ks * 32;
        // A fragment: lanes 0-15 hold K pairs {0,2,4,6,16,18,20,22}; lanes 16-31 +8
        Frag16 a;
        const int kh = hi * 8;
        #pragma unroll
        for (int i = 0; i < 8; i++) {
            int kk = k0 + kh + ((i < 4) ? 2 * i : 16 + 2 * (i - 4));
            a.u[i] = arow[kk >> 1];
        }
        #pragma unroll
        for (int ct = 0; ct < 8; ct++) {
            // B fragment: lane = column n; lanes 0-15 K pairs {0,2,4,6,16,18,20,22}; lanes 16-31 +8
            const uint32_t* bcol = (const uint32_t*)(Bt + (size_t)(ct * 16 + m) * 128);
            Frag16 b;
            const int kb = hi * 8;
            #pragma unroll
            for (int i = 0; i < 8; i++) {
                int kk = k0 + kb + ((i < 4) ? 2 * i : 16 + 2 * (i - 4));
                b.u[i] = bcol[kk >> 1];
            }
            acc[ct] = __builtin_amdgcn_wmma_f32_16x16x32_bf16(
                false, a.v, false, b.v, (short)0, acc[ct], false, false);
        }
    }

    // C/D layout: VGPR v -> M = v + hi*8 ; column N = lane&15
    #pragma unroll
    for (int ct = 0; ct < 8; ct++) {
        #pragma unroll
        for (int v = 0; v < 8; v++) {
            int orow = row_base + v + hi * 8;
            if (orow < nrows) {
                int col = ct * 16 + m;
                float val = acc[ct][v];
                if (bias) val += bias[col];
                C[(size_t)orow * 128 + col] = val;
            }
        }
    }
}

// ---------------- per-relation scatter: out[dst] += hr[src] / max(cnt,1) ----------------
// one wave (32 lanes) per edge, 4 floats per lane

__global__ __launch_bounds__(256) void scatter_kernel(const int* __restrict__ src,
                                                      const int* __restrict__ dst,
                                                      const int* __restrict__ et,
                                                      const float* __restrict__ hr,
                                                      const float* __restrict__ cnt,
                                                      float* __restrict__ out,
                                                      int E, int r, int N) {
    int wid  = (blockIdx.x * 256 + threadIdx.x) >> 5;
    int lane = threadIdx.x & 31;
    if (wid >= E) return;
    if (et[wid] != r) return;                    // wave-uniform
    int s = src[wid], d = dst[wid];
    float c   = cnt[(size_t)r * N + d];
    float inv = 1.0f / fmaxf(c, 1.0f);
    const float4* hv = (const float4*)(hr + (size_t)s * 128);
    float4 v = hv[lane];
    float* o = out + (size_t)d * 128 + lane * 4;
    unsafeAtomicAdd(o + 0, v.x * inv);
    unsafeAtomicAdd(o + 1, v.y * inv);
    unsafeAtomicAdd(o + 2, v.z * inv);
    unsafeAtomicAdd(o + 3, v.w * inv);
}

__global__ __launch_bounds__(256) void relu_kernel(float* __restrict__ out, int n) {
    int i = blockIdx.x * 256 + threadIdx.x;
    if (i < n) out[i] = fmaxf(out[i], 0.0f);
}

// ---------------- host side ----------------

extern "C" void kernel_launch(void* const* d_in, const int* in_sizes, int n_in,
                              void* d_out, int out_size, void* d_ws, size_t ws_size,
                              hipStream_t stream) {
    const float* x          = (const float*)d_in[0];
    const int*   edge_index = (const int*)d_in[1];
    const int*   edge_type  = (const int*)d_in[2];
    const float* W          = (const float*)d_in[3];
    const float* root       = (const float*)d_in[4];
    const float* bias       = (const float*)d_in[5];

    const int N = in_sizes[0] / 128;     // 100000
    const int E = in_sizes[2];           // 600000
    const int NREL = in_sizes[3] / (128 * 128);  // 8
    const int* src = edge_index;
    const int* dst = edge_index + E;
    float* out = (float*)d_out;

    // workspace carve-out (256B aligned slices)
    char* ws = (char*)d_ws;
    auto take = [&](size_t bytes) -> char* {
        char* p = ws;
        ws += (bytes + 255) & ~(size_t)255;
        return p;
    };
    unsigned short* xbf = (unsigned short*)take((size_t)N * 128 * 2);
    unsigned short* wt  = (unsigned short*)take((size_t)NREL * 16384 * 2);
    unsigned short* rt  = (unsigned short*)take((size_t)16384 * 2);
    float*          cnt = (float*)take((size_t)NREL * N * 4);
    float*          hr  = (float*)take((size_t)N * 128 * 4);

    const int NX = N * 128;
    convx_kernel<<<(NX + 255) / 256, 256, 0, stream>>>(x, xbf, NX);
    convw_kernel<<<(NREL * 16384 + 255) / 256, 256, 0, stream>>>(W, wt, NREL);
    convw_kernel<<<(16384 + 255) / 256, 256, 0, stream>>>(root, rt, 1);
    zero_kernel<<<(NREL * N + 255) / 256, 256, 0, stream>>>(cnt, NREL * N);
    count_kernel<<<(E + 255) / 256, 256, 0, stream>>>(edge_type, dst, cnt, E, N);

    const int gemm_blocks = (N + 127) / 128;   // 128 rows per block (8 waves x 16 rows)
    // base: out = x @ root + bias
    rgcn_gemm_bf16<<<gemm_blocks, 256, 0, stream>>>(xbf, rt, bias, out, N);

    const int scat_blocks = (int)(((size_t)E * 32 + 255) / 256);
    for (int r = 0; r < NREL; r++) {
        rgcn_gemm_bf16<<<gemm_blocks, 256, 0, stream>>>(xbf, wt + (size_t)r * 16384,
                                                        nullptr, hr, N);
        scatter_kernel<<<scat_blocks, 256, 0, stream>>>(src, dst, edge_type, hr, cnt,
                                                        out, E, r, N);
    }

    relu_kernel<<<(NX + 255) / 256, 256, 0, stream>>>(out, NX);
}
